// SparseMoE_73443940761663
// MI455X (gfx1250) — matrix operand order repaired and checked by reading, not compiled, not run on hardware
//
#include <hip/hip_runtime.h>

typedef __attribute__((ext_vector_type(16))) __bf16 v16bf;
typedef __attribute__((ext_vector_type(8)))  __bf16 v8bf;
typedef __attribute__((ext_vector_type(4)))  __bf16 v4bf;
typedef __attribute__((ext_vector_type(8)))  float  v8f;
typedef __attribute__((ext_vector_type(4)))  float  v4f;

namespace {
constexpr int kB = 2, kS = 2048, kE = 1024, kN = 8, kH = 4096;
constexpr int kT = kB * kS;              // 4096 tokens
constexpr int BM = 128, BN = 128, BK = 32;
constexpr int LDT = BK + 8;              // LDS row stride (bf16 elems), pad kills bank conflicts
}

// ---------------------------------------------------------------------------
// fp32 -> bf16 straight conversion (x)
// ---------------------------------------------------------------------------
__global__ __launch_bounds__(256) void cvt_bf16_kernel(const float* __restrict__ src,
                                                       __bf16* __restrict__ dst, int n4) {
    int i = blockIdx.x * blockDim.x + threadIdx.x;
    if (i < n4) {
        v4f v = *(const v4f*)(src + (size_t)i * 4);
        v4bf o = { (__bf16)v[0], (__bf16)v[1], (__bf16)v[2], (__bf16)v[3] };
        *(v4bf*)(dst + (size_t)i * 4) = o;
    }
}

// ---------------------------------------------------------------------------
// Tiled transpose + fp32->bf16:  dst[c][r] = src[r][c], per-expert slab in z
// ---------------------------------------------------------------------------
__global__ __launch_bounds__(256) void transpose_cvt_kernel(const float* __restrict__ src,
                                                            __bf16* __restrict__ dst,
                                                            int R, int C) {
    __shared__ float tile[32][33];
    const int tx = threadIdx.x & 31;
    const int ty = threadIdx.x >> 5;                 // 0..7
    const int r0 = blockIdx.y * 32, c0 = blockIdx.x * 32;
    const float* s = src + (size_t)blockIdx.z * R * C;
    __bf16* d = dst + (size_t)blockIdx.z * R * C;
#pragma unroll
    for (int i = 0; i < 4; ++i)
        tile[ty + i * 8][tx] = s[(size_t)(r0 + ty + i * 8) * C + (c0 + tx)];
    __syncthreads();
#pragma unroll
    for (int i = 0; i < 4; ++i)
        d[(size_t)(c0 + ty + i * 8) * R + (r0 + tx)] = (__bf16)tile[tx][ty + i * 8];
}

// ---------------------------------------------------------------------------
// Router: logits = x @ Wr + br, keep top-2, softmax over kept -> wts[T][8]
// One wave32 per token.
// ---------------------------------------------------------------------------
__global__ __launch_bounds__(256) void router_kernel(const float* __restrict__ x,
                                                     const float* __restrict__ Wr,
                                                     const float* __restrict__ br,
                                                     float* __restrict__ wts) {
    const int wave = threadIdx.x >> 5;
    const int lane = threadIdx.x & 31;
    const int t = blockIdx.x * 8 + wave;
    if (t >= kT) return;
    const float* xp = x + (size_t)t * kE;

    float acc[kN] = {0.f, 0.f, 0.f, 0.f, 0.f, 0.f, 0.f, 0.f};
    for (int k = lane; k < kE; k += 32) {
        float xv = xp[k];
        v4f w0 = *(const v4f*)(Wr + (size_t)k * kN);
        v4f w1 = *(const v4f*)(Wr + (size_t)k * kN + 4);
#pragma unroll
        for (int n = 0; n < 4; ++n) acc[n]     = fmaf(xv, w0[n], acc[n]);
#pragma unroll
        for (int n = 0; n < 4; ++n) acc[n + 4] = fmaf(xv, w1[n], acc[n + 4]);
    }
#pragma unroll
    for (int off = 16; off > 0; off >>= 1) {
#pragma unroll
        for (int n = 0; n < kN; ++n) acc[n] += __shfl_down(acc[n], off, 32);
    }
    if (lane == 0) {
        float lg[kN];
#pragma unroll
        for (int n = 0; n < kN; ++n) lg[n] = acc[n] + br[n];
        int i0 = 0;
#pragma unroll
        for (int n = 1; n < kN; ++n) if (lg[n] > lg[i0]) i0 = n;
        int i1 = (i0 == 0) ? 1 : 0;
#pragma unroll
        for (int n = 0; n < kN; ++n) if (n != i0 && lg[n] > lg[i1]) i1 = n;
        float m  = lg[i0];
        float e0 = __expf(lg[i0] - m), e1 = __expf(lg[i1] - m);
        float inv = 1.f / (e0 + e1);
#pragma unroll
        for (int n = 0; n < kN; ++n)
            wts[(size_t)t * kN + n] = (n == i0) ? e0 * inv : (n == i1) ? e1 * inv : 0.f;
    }
}

// ---------------------------------------------------------------------------
// bf16 NT-GEMM with WMMA:  C[M x Nc] = A[M x K] * Bt[Nc x K]^T
// EPI=0: out_bf = bf16(gelu(C + bias[col]))          (hidden activations)
// EPI=1: out_f[row,col] += wts[row,expert]*(C + bias[col])   (MoE combine)
// 256 threads = 8 waves (2x4), block tile 128x128, wave tile 64x32, K-step 32.
// ---------------------------------------------------------------------------
template <int EPI>
__global__ __launch_bounds__(256) void gemm_bf16_kernel(const __bf16* __restrict__ A,
                                                        const __bf16* __restrict__ Bt,
                                                        int M, int Nc, int K,
                                                        const float* __restrict__ bias,
                                                        const float* __restrict__ wts,
                                                        int expert,
                                                        __bf16* __restrict__ outBf,
                                                        float* __restrict__ outF) {
    __shared__ __align__(16) __bf16 As[BM * LDT];
    __shared__ __align__(16) __bf16 Bs[BN * LDT];

    const int tid  = threadIdx.x;
    const int m0   = blockIdx.y * BM;
    const int n0   = blockIdx.x * BN;

    // cooperative copy mapping: 512 chunks of 8 bf16 per tile, 2 per thread
    const int r0c = tid >> 2;                    // 0..63
    const int k0c = (tid & 3) * 8;
    const int r1c = (tid + 256) >> 2;            // 64..127
    const int k1c = ((tid + 256) & 3) * 8;

    // wave / fragment mapping
    const int wave = tid >> 5;
    const int lane = tid & 31;
    const int wm   = wave & 1;                   // 2 waves along M
    const int wn   = wave >> 1;                  // 4 waves along N
    const int lr   = lane & 15;                  // row/col within 16-frag
    const int kh   = lane >> 4;                  // K-half select

    v8f acc[4][2] = {};

    for (int k0 = 0; k0 < K; k0 += BK) {
        // global -> regs (contiguous 16B bf16 chunks along K)
        v8bf a0 = *(const v8bf*)(A  + (size_t)(m0 + r0c) * K + k0 + k0c);
        v8bf a1 = *(const v8bf*)(A  + (size_t)(m0 + r1c) * K + k0 + k1c);
        v8bf b0 = *(const v8bf*)(Bt + (size_t)(n0 + r0c) * K + k0 + k0c);
        v8bf b1 = *(const v8bf*)(Bt + (size_t)(n0 + r1c) * K + k0 + k1c);
        if (k0 + BK < K) {                       // -> global_prefetch_b8
            __builtin_prefetch(A  + (size_t)(m0 + r0c) * K + k0 + BK + k0c, 0, 1);
            __builtin_prefetch(Bt + (size_t)(n0 + r0c) * K + k0 + BK + k0c, 0, 1);
        }
        __syncthreads();                         // protect LDS readers of prev tile
        *(v8bf*)(As + r0c * LDT + k0c) = a0;
        *(v8bf*)(As + r1c * LDT + k1c) = a1;
        *(v8bf*)(Bs + r0c * LDT + k0c) = b0;
        *(v8bf*)(Bs + r1c * LDT + k1c) = b1;
        __syncthreads();

        // fragments per ISA 7.12.2 layouts
        v16bf af[4], bfr[2];
#pragma unroll
        for (int i = 0; i < 4; ++i) {
            const __bf16* ap = As + (wm * 64 + i * 16 + lr) * LDT;
            v8bf lo = *(const v8bf*)(ap + kh * 8);        // K 0..7 / 8..15
            v8bf hi = *(const v8bf*)(ap + 16 + kh * 8);   // K 16..23 / 24..31
            af[i] = __builtin_shufflevector(lo, hi, 0,1,2,3,4,5,6,7,8,9,10,11,12,13,14,15);
        }
#pragma unroll
        for (int j = 0; j < 2; ++j) {
            const __bf16* bp = Bs + (wn * 32 + j * 16 + lr) * LDT;
            v8bf lo = *(const v8bf*)(bp + kh * 8);
            v8bf hi = *(const v8bf*)(bp + 16 + kh * 8);
            bfr[j] = __builtin_shufflevector(lo, hi, 0,1,2,3,4,5,6,7,8,9,10,11,12,13,14,15);
        }
#pragma unroll
        for (int i = 0; i < 4; ++i)
#pragma unroll
            for (int j = 0; j < 2; ++j)
                acc[i][j] = __builtin_amdgcn_wmma_f32_16x16x32_bf16(
                    false, af[i], false, bfr[j], (short)0, acc[i][j], false, false);
    }

    // epilogue (C layout: lane<16 -> N=lane, M=vgpr r; lane>=16 -> M=r+8)
#pragma unroll
    for (int i = 0; i < 4; ++i) {
#pragma unroll
        for (int j = 0; j < 2; ++j) {
            const int col     = n0 + wn * 32 + j * 16 + lr;
            const int rowBase = m0 + wm * 64 + i * 16 + kh * 8;
            const float bc    = bias[col];
#pragma unroll
            for (int r = 0; r < 8; ++r) {
                const int row = rowBase + r;
                float v = acc[i][j][r] + bc;
                if (EPI == 0) {
                    // exact GELU: x * 0.5 * (1 + erf(x/sqrt(2)))
                    float g = 0.5f * v * (1.0f + erff(v * 0.70710678118654752f));
                    outBf[(size_t)row * Nc + col] = (__bf16)g;
                } else {
                    float w = wts[(size_t)row * kN + expert];
                    outF[(size_t)row * Nc + col] += w * v;
                }
            }
        }
    }
}

// ---------------------------------------------------------------------------
// Host launch
// ---------------------------------------------------------------------------
extern "C" void kernel_launch(void* const* d_in, const int* in_sizes, int n_in,
                              void* d_out, int out_size, void* d_ws, size_t ws_size,
                              hipStream_t stream) {
    (void)in_sizes; (void)n_in; (void)out_size; (void)ws_size;
    const float* x  = (const float*)d_in[0];   // [T, E]
    const float* Wr = (const float*)d_in[1];   // [E, N]
    const float* br = (const float*)d_in[2];   // [N]
    const float* W1 = (const float*)d_in[3];   // [N, E, H]
    const float* b1 = (const float*)d_in[4];   // [N, H]
    const float* W2 = (const float*)d_in[5];   // [N, H, E]
    const float* b2 = (const float*)d_in[6];   // [N, E]
    float* out = (float*)d_out;                // [T, E]

    // workspace layout (bytes, 256B aligned)
    char* ws = (char*)d_ws;
    const size_t XB_BYTES  = (size_t)kT * kE * 2;            //   8 MB
    const size_t W1T_BYTES = (size_t)kN * kH * kE * 2;       //  64 MB
    const size_t W2T_BYTES = (size_t)kN * kE * kH * 2;       //  64 MB
    const size_t HID_BYTES = (size_t)kT * kH * 2;            //  32 MB
    __bf16* Xb   = (__bf16*)(ws);
    __bf16* W1t  = (__bf16*)(ws + XB_BYTES);                 // [n][H][E]
    __bf16* W2t  = (__bf16*)(ws + XB_BYTES + W1T_BYTES);     // [n][E][H]
    __bf16* hid  = (__bf16*)(ws + XB_BYTES + W1T_BYTES + W2T_BYTES);
    float*  wts  = (float*)(ws + XB_BYTES + W1T_BYTES + W2T_BYTES + HID_BYTES);

    hipMemsetAsync(out, 0, (size_t)kT * kE * sizeof(float), stream);

    // 1) x -> bf16
    {
        int n4 = kT * kE / 4;
        cvt_bf16_kernel<<<(n4 + 255) / 256, 256, 0, stream>>>(x, Xb, n4);
    }
    // 2) W1[n] (E x H) -> W1t[n] (H x E) bf16 ; W2[n] (H x E) -> W2t[n] (E x H) bf16
    transpose_cvt_kernel<<<dim3(kH / 32, kE / 32, kN), 256, 0, stream>>>(W1, W1t, kE, kH);
    transpose_cvt_kernel<<<dim3(kE / 32, kH / 32, kN), 256, 0, stream>>>(W2, W2t, kH, kE);
    // 3) router weights
    router_kernel<<<kT / 8, 256, 0, stream>>>(x, Wr, br, wts);

    // 4) per-expert fused FFN: hid = gelu(X@W1+b1); out += w * (hid@W2+b2)
    for (int n = 0; n < kN; ++n) {
        gemm_bf16_kernel<0><<<dim3(kH / BN, kT / BM), 256, 0, stream>>>(
            Xb, W1t + (size_t)n * kH * kE, kT, kH, kE,
            b1 + (size_t)n * kH, nullptr, n, hid, nullptr);
        gemm_bf16_kernel<1><<<dim3(kE / BN, kT / BM), 256, 0, stream>>>(
            hid, W2t + (size_t)n * kE * kH, kT, kE, kH,
            b2 + (size_t)n * kE, wts, n, nullptr, out);
    }
}
